// MobiuAttentionFast_77094662963484
// MI455X (gfx1250) — compile-verified, matrix operand order checked by reading, not run on hardware
//
#include <hip/hip_runtime.h>
#include <hip/hip_bf16.h>

// ---------------------------------------------------------------------------
// MobiuAttention for MI455X (gfx1250, wave32, WMMA).
//   1) cvt: x -> f16; weights -> f16 transposed [N,K] (q|k|v stacked)
//   2) wmma_gemm (v_wmma_f32_16x16x32_f16):
//        qkv[M,1536] = x @ [Wq|Wk|Wv], elu+1 on cols < 1024 (q,k)
//        h1[M,128]   = tanh(x @ Ws1 + bs1)
//   3) sensor2: lc = sigmoid(h1 @ Ws2 + bs2)
//   4) chunked linear scan (the step operator S <- diag(lam) S + k (x) v is
//      linear, so a 64-step chunk is S_in -> prod(lam) (.) S_in + S_loc):
//        passA (256 blk): per-chunk local state / decay product / k-sum
//        passB ( 16 blk): sequential 16-chunk combine -> per-chunk init state
//        passC (256 blk): replay chunk from init state, emit y and qks
//      State 64x64 lives in VGPRs (16 per thread); LDS double-buffered
//      staging gives ONE s_barrier per scan step.
//   5) norm: z = sum_e qks via shfl_xor wave reduce; yn = f16(y/z)
//   6) wmma_gemm: out = yn @ Wo (f32)
// ---------------------------------------------------------------------------

typedef __attribute__((ext_vector_type(16))) _Float16 v16h;
typedef __attribute__((ext_vector_type(8)))  _Float16 v8h;
typedef __attribute__((ext_vector_type(8)))  float    v8f;

#define BB  2
#define TTT 1024
#define DDD 512
#define HHH 8
#define EEE 64
#define DD4 128
#define MMM (BB * TTT)     // 2048
#define NQKV (3 * DDD)     // 1536
#define NCH 16             // chunks per (b,h)
#define CLEN 64            // steps per chunk

// ---------------------------------------------------------------------------
// 16x32 f16 fragment load (A/B of wmma.f32.16x16x32), row-major src, ld elems.
// CDNA5 striping: lane<16 holds K k0+0..7 & k0+16..23; lane>=16 holds
// K k0+8..15 & k0+24..31  => two contiguous 16B loads per lane.
// ---------------------------------------------------------------------------
__device__ __forceinline__ v16h load_frag16x32(const _Float16* __restrict__ p,
                                               int row0, int ld, int k0) {
  const int lane = threadIdx.x & 31;
  const int r    = lane & 15;
  const int hi   = lane >> 4;
  const _Float16* base = p + (size_t)(row0 + r) * ld + k0 + hi * 8;
  v8h lo = *(const v8h*)(base);
  v8h hp = *(const v8h*)(base + 16);
  v16h f;
#pragma unroll
  for (int i = 0; i < 8; ++i) { f[i] = lo[i]; f[i + 8] = hp[i]; }
  return f;
}

// ---------------------------------------------------------------------------
// WMMA GEMM: C[M,N] = act(A[M,K] @ Bt[N,K]^T + bias)
// 256 thr = 8 waves; block tile 128x128; wave tile 32x64 => 8 wmma / K-step.
// act: 0 none, 1 elu(x)+1 on cols < actlim, 2 tanh
// ---------------------------------------------------------------------------
__global__ __launch_bounds__(256) void mobiu_wmma_gemm(
    const _Float16* __restrict__ A, const _Float16* __restrict__ Bt,
    const float* __restrict__ bias, float* __restrict__ C,
    int Ndim, int Kdim, int act, int actlim) {
  const int wave = threadIdx.x >> 5;
  const int lane = threadIdx.x & 31;
  const int wm   = wave & 3;
  const int wn   = wave >> 2;
  const int m0   = blockIdx.y * 128 + wm * 32;
  const int n0   = blockIdx.x * 128 + wn * 64;

  v8f zero = {};
  v8f acc[2][4];
#pragma unroll
  for (int i = 0; i < 2; ++i)
#pragma unroll
    for (int j = 0; j < 4; ++j) acc[i][j] = zero;

  for (int k0 = 0; k0 < Kdim; k0 += 32) {
    v16h a[2];
    v16h b[4];
    a[0] = load_frag16x32(A, m0,      Kdim, k0);
    a[1] = load_frag16x32(A, m0 + 16, Kdim, k0);
#pragma unroll
    for (int j = 0; j < 4; ++j)
      b[j] = load_frag16x32(Bt, n0 + 16 * j, Kdim, k0);
#pragma unroll
    for (int i = 0; i < 2; ++i)
#pragma unroll
      for (int j = 0; j < 4; ++j)
        acc[i][j] = __builtin_amdgcn_wmma_f32_16x16x32_f16(
            false, a[i], false, b[j], (short)0, acc[i][j], false, false);
  }

  // C/D layout: VGPR g, lane<16 -> (M=g,N=lane); lane>=16 -> (M=g+8,N=lane-16)
  const int r  = lane & 15;
  const int hi = lane >> 4;
#pragma unroll
  for (int i = 0; i < 2; ++i) {
#pragma unroll
    for (int j = 0; j < 4; ++j) {
      const int col  = n0 + j * 16 + r;
      const float bv = bias ? bias[col] : 0.0f;
      float* cp = C + (size_t)(m0 + i * 16 + hi * 8) * Ndim + col;
#pragma unroll
      for (int g = 0; g < 8; ++g) {
        float vacc = acc[i][j][g] + bv;
        if (act == 1 && col < actlim)
          vacc = (vacc > 0.0f) ? (vacc + 1.0f) : __expf(vacc);   // elu+1
        else if (act == 2)
          vacc = tanhf(vacc);
        cp[(size_t)g * Ndim] = vacc;
      }
    }
  }
}

// ---------------------------------------------------------------------------
// Conversions / prep
// ---------------------------------------------------------------------------
__global__ void mobiu_cvt_f16(const float* __restrict__ in,
                              _Float16* __restrict__ out, int n) {
  int i = blockIdx.x * blockDim.x + threadIdx.x;
  if (i < n) out[i] = (_Float16)in[i];
}

__global__ void mobiu_cvt_wT(const float* __restrict__ W,     // [K,N]
                             _Float16* __restrict__ Wt,       // [N,K]
                             int K, int N) {
  int i = blockIdx.x * blockDim.x + threadIdx.x;
  if (i >= K * N) return;
  int kk = i / N, nn = i - kk * N;
  Wt[(size_t)nn * K + kk] = (_Float16)W[i];
}

__global__ void mobiu_prep_decay(const float* __restrict__ dp,
                                 float* __restrict__ bd, int n) {
  int i = blockIdx.x * blockDim.x + threadIdx.x;
  if (i < n) bd[i] = 1.0f / (1.0f + __expf(-dp[i]));
}

__global__ void mobiu_sensor2(const float* __restrict__ h1,   // [M,128]
                              const float* __restrict__ Ws2,  // [128,8]
                              const float* __restrict__ bs2,  // [8]
                              float* __restrict__ lc) {       // [M,8]
  int idx = blockIdx.x * blockDim.x + threadIdx.x;
  if (idx >= MMM * HHH) return;
  int m = idx >> 3, h = idx & 7;
  float acc = bs2[h];
  const float* row = h1 + (size_t)m * DD4;
#pragma unroll 4
  for (int k = 0; k < DD4; ++k) acc = fmaf(row[k], Ws2[k * HHH + h], acc);
  lc[idx] = 1.0f / (1.0f + __expf(-acc));
}

__device__ __forceinline__ float mobiu_lam(float bde, float lcv) {
  float l = bde * (1.0f + 0.2f * lcv);
  return fminf(fmaxf(l, 0.01f), 0.9995f);
}

// ---------------------------------------------------------------------------
// Pass A: chunk-local scan from zero state. Block = (bh, chunk); thread
// (g=tid>>6, f=tid&63) owns S[16g..16g+15][f]. One barrier per step via
// double-buffered LDS staging.
// ---------------------------------------------------------------------------
__global__ __launch_bounds__(256) void mobiu_scan_passA(
    const float* __restrict__ qkv,   // [M,1536] (q|k|v, head-major)
    const float* __restrict__ lc,    // [M,8]
    const float* __restrict__ bd,    // [8,64]
    float* __restrict__ Sloc,        // [16*16][64][64]
    float* __restrict__ lamprod,     // [16*16][64]
    float* __restrict__ ksumloc) {   // [16*16][64]
  const int bc = blockIdx.x;          // bh*16 + chunk
  const int bh = bc >> 4, c = bc & 15;
  const int b  = bh >> 3, h = bh & 7;
  const int t  = threadIdx.x, f = t & 63, g = t >> 6;

  __shared__ float k_s[2][EEE], v_s[2][EEE], lam_s[2][EEE];

  float S[16];
#pragma unroll
  for (int i = 0; i < 16; ++i) S[i] = 0.0f;
  float lamp = 1.0f, kacc = 0.0f;               // g==0 lanes (e = f)
  const float bde = bd[h * EEE + f];

  for (int s = 0; s < CLEN; ++s) {
    const int m   = b * TTT + c * CLEN + s;
    const int row = m * NQKV + h * EEE;
    const int db  = s & 1;
    if (g == 0) {
      k_s[db][f]   = qkv[row + DDD + f];
      lam_s[db][f] = mobiu_lam(bde, lc[m * HHH + h]);
    } else if (g == 1) {
      v_s[db][f] = qkv[row + 2 * DDD + f];
    } else if (g == 2 && s + 1 < CLEN) {
      __builtin_prefetch(qkv + row + NQKV + DDD + f, 0, 0);     // next k
    } else if (g == 3 && s + 1 < CLEN) {
      __builtin_prefetch(qkv + row + NQKV + 2 * DDD + f, 0, 0); // next v
    }
    __syncthreads();
    const float vf = v_s[db][f];
#pragma unroll
    for (int i = 0; i < 16; ++i) {
      const int e = g * 16 + i;
      S[i] = fmaf(lam_s[db][e], S[i], k_s[db][e] * vf);
    }
    if (g == 0) { lamp *= lam_s[db][f]; kacc += k_s[db][f]; }
  }

  const size_t base = (size_t)bc * EEE * EEE;
#pragma unroll
  for (int i = 0; i < 16; ++i) Sloc[base + (size_t)(g * 16 + i) * EEE + f] = S[i];
  if (g == 0) { lamprod[bc * EEE + f] = lamp; ksumloc[bc * EEE + f] = kacc; }
}

// ---------------------------------------------------------------------------
// Pass B: sequential combine over the 16 chunks of one (b,h).
// Sinit[j] = state before chunk j; S_run <- lamprod_j (.) S_run + Sloc_j.
// ---------------------------------------------------------------------------
__global__ __launch_bounds__(256) void mobiu_scan_passB(
    const float* __restrict__ Sloc, const float* __restrict__ lamprod,
    const float* __restrict__ ksumloc,
    float* __restrict__ Sinit, float* __restrict__ ksuminit) {
  const int bh = blockIdx.x;
  const int t = threadIdx.x, f = t & 63, g = t >> 6;
  __shared__ float lam_s[EEE];

  float S[16];
#pragma unroll
  for (int i = 0; i < 16; ++i) S[i] = 0.0f;
  float ksum = 0.0f;

  for (int j = 0; j < NCH; ++j) {
    const int bc = bh * NCH + j;
    if (g == 0) lam_s[f] = lamprod[bc * EEE + f];
    __syncthreads();
    const size_t base = (size_t)bc * EEE * EEE;
#pragma unroll
    for (int i = 0; i < 16; ++i) {
      const size_t off = base + (size_t)(g * 16 + i) * EEE + f;
      Sinit[off] = S[i];
      S[i] = fmaf(lam_s[g * 16 + i], S[i], Sloc[off]);
    }
    if (g == 0) { ksuminit[bc * EEE + f] = ksum; ksum += ksumloc[bc * EEE + f]; }
    __syncthreads();
  }
}

// ---------------------------------------------------------------------------
// Pass C: replay each chunk from its initial state, emit y (unnormalized) and
// qks = q_e * cumsum(k)_e. One barrier per step: the y-flush for step s-1 is
// pipelined right after the barrier of step s (barrier makes psum[s-1] valid).
// ---------------------------------------------------------------------------
__global__ __launch_bounds__(256) void mobiu_scan_passC(
    const float* __restrict__ qkv, const float* __restrict__ lc,
    const float* __restrict__ bd,
    const float* __restrict__ Sinit, const float* __restrict__ ksuminit,
    float* __restrict__ y,          // [M,512]
    float* __restrict__ qks) {      // [M,512]
  const int bc = blockIdx.x;
  const int bh = bc >> 4, c = bc & 15;
  const int b  = bh >> 3, h = bh & 7;
  const int t  = threadIdx.x, f = t & 63, g = t >> 6;

  __shared__ float q_s[2][EEE], k_s[2][EEE], v_s[2][EEE], lam_s[2][EEE];
  __shared__ float psum[2][256];

  const size_t base = (size_t)bc * EEE * EEE;
  float S[16];
#pragma unroll
  for (int i = 0; i < 16; ++i) S[i] = Sinit[base + (size_t)(g * 16 + i) * EEE + f];
  float ksum = (g == 0) ? ksuminit[bc * EEE + f] : 0.0f;
  const float bde = bd[h * EEE + f];

  int prev_row = 0;
  for (int s = 0; s < CLEN; ++s) {
    const int m      = b * TTT + c * CLEN + s;
    const int in_row = m * NQKV + h * EEE;
    const int row    = m * DDD + h * EEE;
    const int db     = s & 1;
    if (g == 0) {
      q_s[db][f]   = qkv[in_row + f];
      k_s[db][f]   = qkv[in_row + DDD + f];
      lam_s[db][f] = mobiu_lam(bde, lc[m * HHH + h]);
    } else if (g == 1) {
      v_s[db][f] = qkv[in_row + 2 * DDD + f];
    } else if (g == 2 && s + 1 < CLEN) {
      __builtin_prefetch(qkv + in_row + NQKV + f, 0, 0);
      __builtin_prefetch(qkv + in_row + NQKV + DDD + f, 0, 0);
    } else if (g == 3 && s + 1 < CLEN) {
      __builtin_prefetch(qkv + in_row + NQKV + 2 * DDD + f, 0, 0);
    }
    __syncthreads();

    if (g == 0 && s > 0) {        // flush y for step s-1 (psum[db^1] valid now)
      const float* pp = psum[db ^ 1];
      y[prev_row + f] = pp[f] + pp[64 + f] + pp[128 + f] + pp[192 + f];
    }

    const float vf = v_s[db][f];
    float acc = 0.0f;
#pragma unroll
    for (int i = 0; i < 16; ++i) {
      const int e = g * 16 + i;
      float sv = fmaf(lam_s[db][e], S[i], k_s[db][e] * vf);
      S[i] = sv;
      acc = fmaf(q_s[db][e], sv, acc);
    }
    psum[db][t] = acc;
    if (g == 0) {
      ksum += k_s[db][f];
      qks[row + f] = q_s[db][f] * ksum;
    }
    prev_row = row;
  }
  __syncthreads();
  if (g == 0) {                   // flush last step (s=63 -> db=1)
    const float* pp = psum[(CLEN - 1) & 1];
    y[prev_row + f] = pp[f] + pp[64 + f] + pp[128 + f] + pp[192 + f];
  }
}

// ---------------------------------------------------------------------------
// z reduce (shfl_xor within wave => ds/permlane path) + normalize + f16 cvt
// ---------------------------------------------------------------------------
__global__ __launch_bounds__(256) void mobiu_norm(
    const float* __restrict__ y, const float* __restrict__ qks,
    _Float16* __restrict__ yn) {
  const int m    = blockIdx.x;
  const int t    = threadIdx.x;
  const int wave = t >> 5;
  const int lane = t & 31;

  float a = qks[(size_t)m * DDD + wave * EEE + lane] +
            qks[(size_t)m * DDD + wave * EEE + 32 + lane];
#pragma unroll
  for (int off = 16; off > 0; off >>= 1) a += __shfl_xor(a, off, 32);

  __shared__ float z_s[HHH];
  if (lane == 0) z_s[wave] = fmaxf(a, 1e-6f);
  __syncthreads();

  for (int c = t; c < DDD; c += 256)
    yn[(size_t)m * DDD + c] = (_Float16)(y[(size_t)m * DDD + c] / z_s[c >> 6]);
}

// ---------------------------------------------------------------------------
// Host orchestration
// ---------------------------------------------------------------------------
extern "C" void kernel_launch(void* const* d_in, const int* in_sizes, int n_in,
                              void* d_out, int out_size, void* d_ws, size_t ws_size,
                              hipStream_t stream) {
  (void)in_sizes; (void)n_in; (void)out_size; (void)ws_size;

  const float* x   = (const float*)d_in[0];
  const float* Wq  = (const float*)d_in[1];
  const float* Wk  = (const float*)d_in[2];
  const float* Wv  = (const float*)d_in[3];
  const float* Wo  = (const float*)d_in[4];
  const float* dp  = (const float*)d_in[5];
  const float* Ws1 = (const float*)d_in[6];
  const float* bs1 = (const float*)d_in[7];
  const float* Ws2 = (const float*)d_in[8];
  const float* bs2 = (const float*)d_in[9];
  float* out = (float*)d_out;

  char* w = (char*)d_ws;
  _Float16* xh     = (_Float16*)w; w += (size_t)MMM * DDD * 2;
  _Float16* WqkvT  = (_Float16*)w; w += (size_t)NQKV * DDD * 2;  // q|k|v stacked
  _Float16* WoT    = (_Float16*)w; w += (size_t)DDD * DDD * 2;
  _Float16* Ws1T   = (_Float16*)w; w += (size_t)DD4 * DDD * 2;
  float* bdv   = (float*)w; w += (size_t)HHH * EEE * 4;
  float* lcb   = (float*)w; w += (size_t)MMM * HHH * 4;
  float* h1    = (float*)w; w += (size_t)MMM * DD4 * 4;
  float* qkv   = (float*)w; w += (size_t)MMM * NQKV * 4;
  float* yb    = (float*)w; w += (size_t)MMM * DDD * 4;
  float* qks   = (float*)w; w += (size_t)MMM * DDD * 4;
  _Float16* yn = (_Float16*)w; w += (size_t)MMM * DDD * 2;
  float* Sloc     = (float*)w; w += (size_t)BB * HHH * NCH * EEE * EEE * 4;
  float* Sinit    = (float*)w; w += (size_t)BB * HHH * NCH * EEE * EEE * 4;
  float* lamprod  = (float*)w; w += (size_t)BB * HHH * NCH * EEE * 4;
  float* ksumloc  = (float*)w; w += (size_t)BB * HHH * NCH * EEE * 4;
  float* ksuminit = (float*)w; w += (size_t)BB * HHH * NCH * EEE * 4;

  // 1) conversions
  {
    int n = MMM * DDD;
    mobiu_cvt_f16<<<(n + 255) / 256, 256, 0, stream>>>(x, xh, n);
    int nw = DDD * DDD;
    mobiu_cvt_wT<<<(nw + 255) / 256, 256, 0, stream>>>(Wq, WqkvT, DDD, DDD);
    mobiu_cvt_wT<<<(nw + 255) / 256, 256, 0, stream>>>(Wk, WqkvT + (size_t)DDD * DDD, DDD, DDD);
    mobiu_cvt_wT<<<(nw + 255) / 256, 256, 0, stream>>>(Wv, WqkvT + (size_t)2 * DDD * DDD, DDD, DDD);
    mobiu_cvt_wT<<<(nw + 255) / 256, 256, 0, stream>>>(Wo, WoT, DDD, DDD);
    int ns = DDD * DD4;
    mobiu_cvt_wT<<<(ns + 255) / 256, 256, 0, stream>>>(Ws1, Ws1T, DDD, DD4);
    mobiu_prep_decay<<<2, 256, 0, stream>>>(dp, bdv, HHH * EEE);
  }

  // 2) fused qkv projection + sensor layer 1 (WMMA)
  {
    dim3 blk(256);
    dim3 gq(NQKV / 128, MMM / 128);   // 12 x 16
    mobiu_wmma_gemm<<<gq, blk, 0, stream>>>(xh, WqkvT, nullptr, qkv,
                                            NQKV, DDD, 1, 2 * DDD);
    dim3 gs(DD4 / 128, MMM / 128);    // 1 x 16
    mobiu_wmma_gemm<<<gs, blk, 0, stream>>>(xh, Ws1T, bs1, h1, DD4, DDD, 2, 0);
  }

  // 3) sensor layer 2
  mobiu_sensor2<<<(MMM * HHH + 255) / 256, 256, 0, stream>>>(h1, Ws2, bs2, lcb);

  // 4) chunked decay scan
  mobiu_scan_passA<<<BB * HHH * NCH, 256, 0, stream>>>(qkv, lcb, bdv,
                                                       Sloc, lamprod, ksumloc);
  mobiu_scan_passB<<<BB * HHH, 256, 0, stream>>>(Sloc, lamprod, ksumloc,
                                                 Sinit, ksuminit);
  mobiu_scan_passC<<<BB * HHH * NCH, 256, 0, stream>>>(qkv, lcb, bdv,
                                                       Sinit, ksuminit, yb, qks);

  // 5) normalize
  mobiu_norm<<<MMM, 256, 0, stream>>>(yb, qks, yn);

  // 6) output projection (WMMA)
  {
    dim3 blk(256);
    dim3 go(DDD / 128, MMM / 128);
    mobiu_wmma_gemm<<<go, blk, 0, stream>>>(yn, WoT, nullptr, out, DDD, DDD, 0, 0);
  }
}
